// Stage_30545807409338
// MI455X (gfx1250) — compile-verified, matrix operand order checked by reading, not run on hardware
//
#include <hip/hip_runtime.h>
#include <hip/hip_bf16.h>
#include <math.h>

typedef __attribute__((ext_vector_type(16))) _Float16 v16h;
typedef __attribute__((ext_vector_type(8)))  float    v8f;
typedef __attribute__((ext_vector_type(4)))  unsigned int v4u;
typedef __attribute__((ext_vector_type(8)))  int      v8i;
typedef __attribute__((ext_vector_type(4)))  int      v4i;

#if defined(__has_builtin)
#  if __has_builtin(__builtin_amdgcn_tensor_load_to_lds)
#    define HAVE_TDM 1
#  endif
#endif
#ifndef HAVE_TDM
#define HAVE_TDM 0
#endif

#define N_PTS 8192
#define BB    2
#define GG    4096
#define CC    256
#define OO    2
#define NDEPTH 2
#define DDI   512
#define DDS   16
#define DDR   16
#define DCV   4
#define KNN   32
#define QQ    128
#define HH    8
#define FF    1024
#define DH    32

// ---------------- input index map (insertion-order flatten of setup_inputs) ----
#define I_FEAT 0
#define I_QUERY 1
#define I_QPOS 2
#define I_BID 3
#define I_ORDER 4
#define I_INV 5
#define I_NBR 6
#define I_CPE_CONV_W 7
#define I_CPE_CONV_B 8
#define I_CPE_FC1_W 9
#define I_CPE_FC1_B 10
#define I_CPE_LN_G 11
#define I_CPE_LN_B 12
// block params: ln_g,ln_b,in_proj,conv_w,conv_b,x_proj,dt_w,dt_b,A_log,D,out_proj
#define BLK(b,o) (13 + (b)*11 + (o))
#define I_NF_G 35
#define I_NF_B 36
#define D1 37   // dec1: cross(wq,bq,wk,bk,wv,bv,wo,bo)=+0..7, ln1_g=+8, ln1_b=+9,
                // ln3_g=+10, ln3_b=+11, w1=+12, b1=+13, w2=+14, b2=+15,
                // self=+16..23, ln2_g=+24, ln2_b=+25
#define D2 63   // dec2: cross=+0..7, ln1_g=+8, ln1_b=+9, ln3_g=+10, ln3_b=+11,
                // w1=+12, b1=+13, w2=+14, b2=+15

// ============================ device helpers ================================

__device__ __forceinline__ float warp_sum(float v) {
#pragma unroll
  for (int m = 16; m > 0; m >>= 1) v += __shfl_xor(v, m, 32);
  return v;
}
__device__ __forceinline__ float warp_max(float v) {
#pragma unroll
  for (int m = 16; m > 0; m >>= 1) v = fmaxf(v, __shfl_xor(v, m, 32));
  return v;
}
// every thread returns the block-wide result; barrier at end makes sh reusable
__device__ __forceinline__ float block_sum(float v, float* sh) {
  int w = threadIdx.x >> 5, l = threadIdx.x & 31, nw = blockDim.x >> 5;
  v = warp_sum(v);
  if (l == 0) sh[w] = v;
  __syncthreads();
  float r = (l < nw) ? sh[l] : 0.f;
  r = warp_sum(r);
  __syncthreads();
  return r;
}
__device__ __forceinline__ float block_max(float v, float* sh) {
  int w = threadIdx.x >> 5, l = threadIdx.x & 31, nw = blockDim.x >> 5;
  v = warp_max(v);
  if (l == 0) sh[w] = v;
  __syncthreads();
  float r = (l < nw) ? sh[l] : -3.402823466e38f;
  r = warp_max(r);
  __syncthreads();
  return r;
}
__device__ __forceinline__ float gelu_f(float x) {
  float x3 = x * x * x;
  return 0.5f * x * (1.f + tanhf(0.7978845608028654f * (x + 0.044715f * x3)));
}
__device__ __forceinline__ float silu_f(float x) { return x / (1.f + __expf(-x)); }

// A-fragment (and B-fragment in W^T mode): row-major row pointer, K-chunk k0.
// ISA 16-bit A layout: lane half h supplies K in {k0+8h .. k0+8h+7} and
// {k0+16+8h .. k0+16+8h+7}.
__device__ __forceinline__ v16h load_frag_row(const float* __restrict__ row,
                                              int k0, int h) {
  const float* p = row + k0 + 8 * h;
  float4 f0 = *(const float4*)(p);
  float4 f1 = *(const float4*)(p + 4);
  float4 f2 = *(const float4*)(p + 16);
  float4 f3 = *(const float4*)(p + 20);
  v16h r;
  r[0] = (_Float16)f0.x;  r[1] = (_Float16)f0.y;
  r[2] = (_Float16)f0.z;  r[3] = (_Float16)f0.w;
  r[4] = (_Float16)f1.x;  r[5] = (_Float16)f1.y;
  r[6] = (_Float16)f1.z;  r[7] = (_Float16)f1.w;
  r[8] = (_Float16)f2.x;  r[9] = (_Float16)f2.y;
  r[10] = (_Float16)f2.z; r[11] = (_Float16)f2.w;
  r[12] = (_Float16)f3.x; r[13] = (_Float16)f3.y;
  r[14] = (_Float16)f3.z; r[15] = (_Float16)f3.w;
  return r;
}
// B-fragment when B is stored (K,N) row-major: column col, stride ldb per K.
__device__ __forceinline__ v16h load_frag_col(const float* __restrict__ Bp,
                                              int ldb, int k0, int h, int col) {
  const float* p = Bp + (long)(k0 + 8 * h) * ldb + col;
  v16h r;
#pragma unroll
  for (int e = 0; e < 8; ++e) {
    r[e]     = (_Float16)p[(long)e * ldb];
    r[8 + e] = (_Float16)p[(long)(16 + e) * ldb];
  }
  return r;
}

#if HAVE_TDM
// Issue a 2D Tensor-Data-Mover load: tile_y rows of tile_x f32 elements,
// row stride = stride_elems, into LDS byte offset lds_off (row-major packed).
// D# layout per CDNA5 ISA 8.3/8.4 (count=1, data_size=4B, type=2 "image").
// This toolchain's builtin takes 6 args (g0 v4u, g1 v8i, g2 v4i, g3 v4i,
// g4 v8i, cpol) -- groups 2/3 unused for 2D tiles, zero-filled.
__device__ __forceinline__ void tdm_load_2d(unsigned lds_off,
                                            const float* gsrc, int tile_x,
                                            int tile_y, long stride_elems) {
  unsigned long ga = (unsigned long)gsrc;
  v4u g0;
  g0[0] = 1u;                                        // count=1, user mode
  g0[1] = lds_off;                                   // lds_addr
  g0[2] = (unsigned)(ga & 0xffffffffu);              // global_addr[31:0]
  g0[3] = (unsigned)((ga >> 32) & 0x1ffffffu) | (2u << 30);  // [120:64],type=2
  unsigned long st = (unsigned long)stride_elems;
  const unsigned td0 = 1u << 30, td1 = 1u << 30;     // huge dims: no OOB clamp
  v8i g1;
  g1[0] = (int)(2u << 16);                           // data_size=4B, mask=0
  g1[1] = (int)((td0 & 0xffffu) << 16);              // tensor_dim0[15:0]
  g1[2] = (int)((td0 >> 16) | ((td1 & 0xffffu) << 16));
  g1[3] = (int)((td1 >> 16) | ((unsigned)tile_x << 16));  // tile_dim0
  g1[4] = (int)((unsigned)tile_y);                   // tile_dim1, tile_dim2=0
  g1[5] = (int)(st & 0xffffffffu);                   // dim0_stride[31:0]
  g1[6] = (int)((st >> 32) & 0xffffu);               // dim0_stride[47:32]
  g1[7] = 0;
  v4i z4 = {0, 0, 0, 0};
  v8i z8 = {0, 0, 0, 0, 0, 0, 0, 0};
  __builtin_amdgcn_tensor_load_to_lds(g0, g1, z4, z4, z8, 0);
}
#endif

// ============================ generic WMMA GEMM =============================
// D[M,N] = act(alpha * A' @ B' + bias) (+= if accum)
//   A' rows optionally gathered through idxA (stride idxStride per row).
//   BT=1: B stored (N,K) row-major (i.e. computes A @ B.T)  -> weights
//   BT=0: B stored (K,N) row-major (plain A @ B)            -> attn A*V etc.
//   NT  : 16-column tiles per wave (register blocking; A fragment reused NT x,
//         NT independent back-to-back WMMAs avoid D->A/B hazard NOPs).
// Batched over grid.z: z=(zb,zh), offsets zb*s?b + zh*s?h for A,B,C.
// Requirements: M%128==0, N%(16*NT)==0, K%32==0. Block=256 (8 waves, 16 rows ea).
template <int BT, int NT>
__global__ __launch_bounds__(256) void k_gemm(
    const float* __restrict__ A, int lda, long sAb, long sAh,
    const int* __restrict__ idxA, int idxStride,
    const float* __restrict__ Bp, int ldb, long sBb, long sBh,
    const float* __restrict__ bias,
    float* __restrict__ Cp, int ldc, long sCb, long sCh,
    int Kdim, int Hnum, float alpha, int act, int accum) {
  int wave = threadIdx.x >> 5;
  int lane = threadIdx.x & 31;
  int h = lane >> 4;
  int ln = lane & 15;
  int zb = blockIdx.z / Hnum, zh = blockIdx.z % Hnum;
  const float* Az = A + zb * sAb + zh * sAh;
  const float* Bz = Bp + zb * sBb + zh * sBh;
  float* Cz = Cp + zb * sCb + zh * sCh;

  int row0 = (blockIdx.x * 8 + wave) * 16;
  int col0 = blockIdx.y * (16 * NT);

  long arow = row0 + ln;
  if (idxA) arow = idxA[(long)(row0 + ln) * idxStride];
  const float* Arow = Az + arow * (long)lda;

  v8f c[NT];
#pragma unroll
  for (int t = 0; t < NT; ++t) c[t] = (v8f){};

  for (int k0 = 0; k0 < Kdim; k0 += 32) {
    if (k0 + 32 < Kdim) __builtin_prefetch(Arow + k0 + 32, 0, 1);
    v16h a = load_frag_row(Arow, k0, h);
    v16h b[NT];
#pragma unroll
    for (int t = 0; t < NT; ++t) {
      if (BT)
        b[t] = load_frag_row(Bz + (long)(col0 + 16 * t + ln) * ldb, k0, h);
      else
        b[t] = load_frag_col(Bz, ldb, k0, h, col0 + 16 * t + ln);
    }
#pragma unroll
    for (int t = 0; t < NT; ++t)
      c[t] = __builtin_amdgcn_wmma_f32_16x16x32_f16(false, a, false, b[t],
                                                    (short)0, c[t], false,
                                                    false);
  }
#pragma unroll
  for (int t = 0; t < NT; ++t) {
#pragma unroll
    for (int r = 0; r < 8; ++r) {
      int row = row0 + r + 8 * h;
      int col = col0 + 16 * t + ln;
      float v = alpha * c[t][r];
      if (bias) v += bias[col];
      if (act == 1) v = gelu_f(v);
      else if (act == 2) v = silu_f(v);
      float* dst = &Cz[(long)row * ldc + col];
      if (accum) v += *dst;
      *dst = v;
    }
  }
}

#if HAVE_TDM
// ===================== TDM-staged WMMA GEMM (NT=4) ==========================
// The block's B tile (64 output columns) is identical across all 8 waves, so
// wave 0 DMAs it into LDS with the Tensor Data Mover (double-buffered K=64
// chunks, per-wave TENSORcnt wait + workgroup barrier to publish), cutting
// global B traffic 8x. A is streamed per-wave from global as before.
// Requirements: M%128==0, N%64==0, K%64==0.
template <int BT>
__global__ __launch_bounds__(256) void k_gemm_tdm(
    const float* __restrict__ A, int lda, long sAb, long sAh,
    const int* __restrict__ idxA, int idxStride,
    const float* __restrict__ Bp, int ldb, long sBb, long sBh,
    const float* __restrict__ bias,
    float* __restrict__ Cp, int ldc, long sCb, long sCh,
    int Kdim, int Hnum, float alpha, int act, int accum) {
  const int KS = 64;                    // K-chunk staged per buffer
  __shared__ float sB[2][64 * KS];      // 2 x 16 KB
  int wave = threadIdx.x >> 5;
  int lane = threadIdx.x & 31;
  int h = lane >> 4;
  int ln = lane & 15;
  int zb = blockIdx.z / Hnum, zh = blockIdx.z % Hnum;
  const float* Az = A + zb * sAb + zh * sAh;
  const float* Bz = Bp + zb * sBb + zh * sBh;
  float* Cz = Cp + zb * sCb + zh * sCh;

  int row0 = (blockIdx.x * 8 + wave) * 16;
  int col0 = blockIdx.y * 64;

  long arow = row0 + ln;
  if (idxA) arow = idxA[(long)(row0 + ln) * idxStride];
  const float* Arow = Az + arow * (long)lda;

  v8f c[4];
#pragma unroll
  for (int t = 0; t < 4; ++t) c[t] = (v8f){};

  int nch = Kdim / KS;
  if (wave == 0) {
    const float* g = BT ? Bz + (long)col0 * ldb : Bz + col0;
    tdm_load_2d(0u, g, BT ? KS : 64, BT ? 64 : KS, ldb);
  }
  for (int ch = 0; ch < nch; ++ch) {
    int kbase = ch * KS;
    if (wave == 0) {
      if (ch + 1 < nch) {
        int kn = kbase + KS;
        const float* g =
            BT ? Bz + (long)col0 * ldb + kn : Bz + (long)kn * ldb + col0;
        tdm_load_2d((unsigned)(((ch + 1) & 1) * (64 * KS * 4)), g,
                    BT ? KS : 64, BT ? 64 : KS, ldb);
        __builtin_amdgcn_s_wait_tensorcnt(1);  // previous chunk landed
      } else {
        __builtin_amdgcn_s_wait_tensorcnt(0);  // last chunk landed
      }
    }
    __syncthreads();  // publish LDS chunk to all waves
    const float* sb = sB[ch & 1];
#pragma unroll
    for (int kk = 0; kk < KS; kk += 32) {
      v16h a = load_frag_row(Arow, kbase + kk, h);
      v16h b[4];
#pragma unroll
      for (int t = 0; t < 4; ++t) {
        if (BT)
          b[t] = load_frag_row(sb + (16 * t + ln) * KS, kk, h);
        else
          b[t] = load_frag_col(sb, 64, kk, h, 16 * t + ln);
      }
#pragma unroll
      for (int t = 0; t < 4; ++t)
        c[t] = __builtin_amdgcn_wmma_f32_16x16x32_f16(false, a, false, b[t],
                                                      (short)0, c[t], false,
                                                      false);
    }
    __syncthreads();  // all waves done with this buffer before reuse
  }
#pragma unroll
  for (int t = 0; t < 4; ++t) {
#pragma unroll
    for (int r = 0; r < 8; ++r) {
      int row = row0 + r + 8 * h;
      int col = col0 + 16 * t + ln;
      float v = alpha * c[t][r];
      if (bias) v += bias[col];
      if (act == 1) v = gelu_f(v);
      else if (act == 2) v = silu_f(v);
      float* dst = &Cz[(long)row * ldc + col];
      if (accum) v += *dst;
      *dst = v;
    }
  }
}
#endif  // HAVE_TDM

// ============================ small kernels ================================

// LayerNorm over rows of length CC. mode 0: ln(x); 1: res + ln(x); 2: ln(x+res)
__global__ void k_ln(const float* __restrict__ x, const float* __restrict__ res,
                     const float* __restrict__ g, const float* __restrict__ b,
                     float* __restrict__ out, int mode) {
  __shared__ float sh[8];
  long row = blockIdx.x;
  int t = threadIdx.x;
  float v = x[row * CC + t];
  float rv = res ? res[row * CC + t] : 0.f;
  if (mode == 2) v += rv;
  float mean = block_sum(v, sh) * (1.f / CC);
  float d = v - mean;
  float var = block_sum(d * d, sh) * (1.f / CC);
  float y = d * rsqrtf(var + 1e-5f) * g[t] + b[t];
  if (mode == 1) y += rv;
  out[row * CC + t] = y;
}

__global__ void k_softmax(float* __restrict__ x, int len) {
  __shared__ float sh[8];
  float* p = x + (long)blockIdx.x * len;
  float m = -3.402823466e38f;
  for (int i = threadIdx.x; i < len; i += blockDim.x) m = fmaxf(m, p[i]);
  m = block_max(m, sh);
  float s = 0.f;
  for (int i = threadIdx.x; i < len; i += blockDim.x) s += __expf(p[i] - m);
  s = block_sum(s, sh);
  float inv = 1.f / s;
  for (int i = threadIdx.x; i < len; i += blockDim.x)
    p[i] = __expf(p[i] - m) * inv;
}

__global__ void k_add(float* __restrict__ out, const float* __restrict__ a,
                      const float* __restrict__ b, long n) {
  long i = (long)blockIdx.x * blockDim.x + threadIdx.x;
  if (i < n) out[i] = a[i] + b[i];
}
__global__ void k_scale(float* __restrict__ x, float s, long n) {
  long i = (long)blockIdx.x * blockDim.x + threadIdx.x;
  if (i < n) x[i] *= s;
}
__global__ void k_gather(float* __restrict__ dst, const float* __restrict__ src,
                         const int* __restrict__ idx) {
  long r = blockIdx.x;
  dst[r * CC + threadIdx.x] = src[(long)idx[r] * CC + threadIdx.x];
}
__global__ void k_gather_add(float* __restrict__ acc,
                             const float* __restrict__ src,
                             const int* __restrict__ idx) {
  long r = blockIdx.x;
  acc[r * CC + threadIdx.x] += src[(long)idx[r] * CC + threadIdx.x];
}
// bias2[j] = fc1_b[j] + sum_i conv_b[i] * fc1_w[j,i]   (folds (conv+cb)@W^T)
__global__ void k_fuse_bias(const float* __restrict__ cb,
                            const float* __restrict__ fw,
                            const float* __restrict__ fb,
                            float* __restrict__ out) {
  int j = threadIdx.x;
  float s = fb[j];
  for (int i = 0; i < CC; ++i) s += cb[i] * fw[j * CC + i];
  out[j] = s;
}
// causal depthwise conv (DCV taps) with sequence-boundary mask, + bias, silu
__global__ void k_conv_silu(const float* __restrict__ xz,
                            const float* __restrict__ cw,
                            const float* __restrict__ cb,
                            const int* __restrict__ bid,
                            float* __restrict__ xc) {
  long i = (long)blockIdx.x * blockDim.x + threadIdx.x;
  long t = i / DDI;
  int d = (int)(i % DDI);
  int myb = bid[t];
  float acc = 0.f;
#pragma unroll
  for (int s = 0; s < DCV; ++s) {
    long ts = t - s;
    if (ts >= 0 && bid[ts] == myb)
      acc += xz[ts * (2 * DDI) + d] * cw[d * DCV + (DCV - 1 - s)];
  }
  acc += cb[d];
  xc[i] = silu_f(acc);
}
// dt = softplus(dbl[:, :16] @ dt_w.T + dt_b)   (K=16, below WMMA granularity)
__global__ void k_dtproj(const float* __restrict__ dbl,
                         const float* __restrict__ dtw,
                         const float* __restrict__ dtb,
                         float* __restrict__ dt) {
  long i = (long)blockIdx.x * blockDim.x + threadIdx.x;
  long t = i / DDI;
  int d = (int)(i % DDI);
  float s = dtb[d];
#pragma unroll
  for (int j = 0; j < DDR; ++j) s += dbl[t * 48 + j] * dtw[d * DDR + j];
  dt[i] = (s > 20.f) ? s : log1pf(__expf(s));
}
// selective-state scan: thread d owns 16-wide state, sequential over t
__global__ void k_scan(const float* __restrict__ dt,
                       const float* __restrict__ xc,
                       const float* __restrict__ xz,  // z at col DDI.. of 2*DDI
                       const float* __restrict__ dbl,
                       const int* __restrict__ bid,
                       const float* __restrict__ A_log,
                       const float* __restrict__ Dp, float* __restrict__ y) {
  int d = blockIdx.x * blockDim.x + threadIdx.x;
  if (d >= DDI) return;
  float Av[DDS], hst[DDS];
#pragma unroll
  for (int s = 0; s < DDS; ++s) {
    Av[s] = -__expf(A_log[d * DDS + s]);
    hst[s] = 0.f;
  }
  float Dd = Dp[d];
  int prev = -1;
  for (long t = 0; t < N_PTS; ++t) {
    int bt = bid[t];
    float keep = (t > 0 && bt == prev) ? 1.f : 0.f;
    prev = bt;
    float dtv = dt[t * DDI + d];
    float xv = xc[t * DDI + d];
    float acc = 0.f;
#pragma unroll
    for (int s = 0; s < DDS; ++s) {
      float Bv = dbl[t * 48 + DDR + s];
      float Cv = dbl[t * 48 + DDR + DDS + s];
      float hs = __expf(dtv * Av[s]) * hst[s] * keep + dtv * xv * Bv;
      hst[s] = hs;
      acc += hs * Cv;
    }
    float zv = xz[t * (2 * DDI) + DDI + d];
    y[t * DDI + d] = (acc + Dd * xv) * silu_f(zv);
  }
}

// ============================ host orchestration ============================

static inline void gemm(hipStream_t st, const float* A, int lda, long sAb,
                        long sAh, const int* idxA, int idxStride,
                        const float* Bp, int ldb, long sBb, long sBh, int bT,
                        const float* bias, float* Cp, int ldc, long sCb,
                        long sCh, int M, int Nn, int Kdim, int nbatch, int Hnum,
                        float alpha, int act, int accum) {
#define GEMM_ARGS A, lda, sAb, sAh, idxA, idxStride, Bp, ldb, sBb, sBh, bias, \
                  Cp, ldc, sCb, sCh, Kdim, Hnum, alpha, act, accum
#if HAVE_TDM
  if (Nn % 64 == 0 && Kdim % 64 == 0) {
    dim3 g(M / 128, Nn / 64, nbatch);
    if (bT) k_gemm_tdm<1><<<g, 256, 0, st>>>(GEMM_ARGS);
    else    k_gemm_tdm<0><<<g, 256, 0, st>>>(GEMM_ARGS);
    return;
  }
#endif
  int nt = (Nn % 64 == 0) ? 4 : (Nn % 32 == 0) ? 2 : 1;
  dim3 g(M / 128, Nn / (16 * nt), nbatch);
  if (bT) {
    if (nt == 4)      k_gemm<1, 4><<<g, 256, 0, st>>>(GEMM_ARGS);
    else if (nt == 2) k_gemm<1, 2><<<g, 256, 0, st>>>(GEMM_ARGS);
    else              k_gemm<1, 1><<<g, 256, 0, st>>>(GEMM_ARGS);
  } else {
    if (nt == 4)      k_gemm<0, 4><<<g, 256, 0, st>>>(GEMM_ARGS);
    else if (nt == 2) k_gemm<0, 2><<<g, 256, 0, st>>>(GEMM_ARGS);
    else              k_gemm<0, 1><<<g, 256, 0, st>>>(GEMM_ARGS);
  }
#undef GEMM_ARGS
}

extern "C" void kernel_launch(void* const* d_in, const int* in_sizes, int n_in,
                              void* d_out, int out_size, void* d_ws,
                              size_t ws_size, hipStream_t stream) {
  (void)in_sizes; (void)n_in; (void)out_size; (void)ws_size;
  const float* feat = (const float*)d_in[I_FEAT];
  const float* query = (const float*)d_in[I_QUERY];
  const float* qpos = (const float*)d_in[I_QPOS];
  const int* bid = (const int*)d_in[I_BID];
  const int* order = (const int*)d_in[I_ORDER];
  const int* inv = (const int*)d_in[I_INV];
  const int* nbr = (const int*)d_in[I_NBR];
#define PF(i) ((const float*)d_in[(i)])

  float* out_feat = (float*)d_out;
  float* out_query = (float*)d_out + (size_t)N_PTS * CC;

  // ---- workspace carve (floats) ----
  float* ws = (float*)d_ws;
  size_t off = 0;
  const size_t S = (size_t)BB * QQ * CC;          // 65536
  const size_t C2 = (size_t)N_PTS * CC;           // 2M
  const size_t BIGN = (size_t)BB * HH * QQ * GG;  // 8.39M
#define ALLOC(n) (ws + (off += (n)) - (n))
  float* s0 = ALLOC(S);  float* s1 = ALLOC(S);  float* s2 = ALLOC(S);
  float* s3 = ALLOC(S);  float* s4 = ALLOC(S);  float* s5 = ALLOC(S);
  float* ffnq = ALLOC((size_t)BB * QQ * FF);
  float* big = ALLOC(BIGN);                       // scores1/xz/scores2/ffn2
  float* c0 = ALLOC(C2);  // kbuf -> seq (r1)
  float* c1 = ALLOC(C2);  // vbuf -> lnbuf / h_end
  float* c2 = ALLOC(C2);  // cpe conv -> r2
  float* c3 = ALLOC(C2);  // cpe x -> mamba_out / mix_out / attn-out dec2
  float* c4 = ALLOC(C2);  // h (cpe output) -> x dec2
  float* c5 = ALLOC(C2);  // acc -> h2
  float* c6 = ALLOC(C2);  // qproj2 / tmp dec2
  float* xcb = ALLOC((size_t)N_PTS * DDI);
  float* dtb = ALLOC((size_t)N_PTS * DDI);
  float* yb = ALLOC((size_t)N_PTS * DDI);
  float* dblb = ALLOC((size_t)N_PTS * 48);
  float* kq2 = ALLOC(S);
  float* vq2 = ALLOC(S);
  float* bias2 = ALLOC(256);

  const float rs = 0.17677669529663687f;  // 1/sqrt(DH)
  const int BQ = BB * QQ;
  const long QC = (long)QQ * CC, GC = (long)GG * CC;

  // ======================= dec1 (self+cross decoder) =======================
  k_add<<<(S + 255) / 256, 256, 0, stream>>>(s0, query, qpos, S);
  // cross attention: q from (query+pos), k/v from feat
  gemm(stream, s0, CC, 0, 0, nullptr, 0, PF(D1 + 0), CC, 0, 0, 1, PF(D1 + 1),
       s1, CC, 0, 0, BQ, CC, CC, 1, 1, 1.f, 0, 0);
  gemm(stream, feat, CC, 0, 0, nullptr, 0, PF(D1 + 2), CC, 0, 0, 1, PF(D1 + 3),
       c0, CC, 0, 0, N_PTS, CC, CC, 1, 1, 1.f, 0, 0);
  gemm(stream, feat, CC, 0, 0, nullptr, 0, PF(D1 + 4), CC, 0, 0, 1, PF(D1 + 5),
       c1, CC, 0, 0, N_PTS, CC, CC, 1, 1, 1.f, 0, 0);
  // scores[b,h] = qh @ kh^T * rs   (M=Q, N=G, K=DH)
  gemm(stream, s1, CC, QC, DH, nullptr, 0, c0, CC, GC, DH, 1, nullptr, big, GG,
       (long)HH * QQ * GG, (long)QQ * GG, QQ, GG, DH, BB * HH, HH, rs, 0, 0);
  k_softmax<<<BB * HH * QQ, 256, 0, stream>>>(big, GG);
  // o[b,h] = probs @ vh   (M=Q, N=DH, K=G)
  gemm(stream, big, GG, (long)HH * QQ * GG, (long)QQ * GG, nullptr, 0, c1, CC,
       GC, DH, 0, nullptr, s2, CC, QC, DH, QQ, DH, GG, BB * HH, HH, 1.f, 0, 0);
  gemm(stream, s2, CC, 0, 0, nullptr, 0, PF(D1 + 6), CC, 0, 0, 1, PF(D1 + 7),
       s5, CC, 0, 0, BQ, CC, CC, 1, 1, 1.f, 0, 0);
  k_ln<<<BQ, 256, 0, stream>>>(s5, query, PF(D1 + 8), PF(D1 + 9), s3, 2);
  // self attention: q,k from (x+pos), v from x
  k_add<<<(S + 255) / 256, 256, 0, stream>>>(s0, s3, qpos, S);
  gemm(stream, s0, CC, 0, 0, nullptr, 0, PF(D1 + 16), CC, 0, 0, 1, PF(D1 + 17),
       s1, CC, 0, 0, BQ, CC, CC, 1, 1, 1.f, 0, 0);
  gemm(stream, s0, CC, 0, 0, nullptr, 0, PF(D1 + 18), CC, 0, 0, 1, PF(D1 + 19),
       kq2, CC, 0, 0, BQ, CC, CC, 1, 1, 1.f, 0, 0);
  gemm(stream, s3, CC, 0, 0, nullptr, 0, PF(D1 + 20), CC, 0, 0, 1, PF(D1 + 21),
       vq2, CC, 0, 0, BQ, CC, CC, 1, 1, 1.f, 0, 0);
  gemm(stream, s1, CC, QC, DH, nullptr, 0, kq2, CC, QC, DH, 1, nullptr, big, QQ,
       (long)HH * QQ * QQ, (long)QQ * QQ, QQ, QQ, DH, BB * HH, HH, rs, 0, 0);
  k_softmax<<<BB * HH * QQ, 128, 0, stream>>>(big, QQ);
  gemm(stream, big, QQ, (long)HH * QQ * QQ, (long)QQ * QQ, nullptr, 0, vq2, CC,
       QC, DH, 0, nullptr, s2, CC, QC, DH, QQ, DH, QQ, BB * HH, HH, 1.f, 0, 0);
  gemm(stream, s2, CC, 0, 0, nullptr, 0, PF(D1 + 22), CC, 0, 0, 1, PF(D1 + 23),
       s5, CC, 0, 0, BQ, CC, CC, 1, 1, 1.f, 0, 0);
  k_ln<<<BQ, 256, 0, stream>>>(s5, s3, PF(D1 + 24), PF(D1 + 25), s4, 2);
  // FFN
  gemm(stream, s4, CC, 0, 0, nullptr, 0, PF(D1 + 12), CC, 0, 0, 1, PF(D1 + 13),
       ffnq, FF, 0, 0, BQ, FF, CC, 1, 1, 1.f, 1, 0);
  gemm(stream, ffnq, FF, 0, 0, nullptr, 0, PF(D1 + 14), FF, 0, 0, 1,
       PF(D1 + 15), s5, CC, 0, 0, BQ, CC, FF, 1, 1, 1.f, 0, 0);
  k_ln<<<BQ, 256, 0, stream>>>(s5, s4, PF(D1 + 10), PF(D1 + 11), out_query, 2);

  // ======================= CPE (gathered WMMA GEMMs) =======================
  (void)hipMemsetAsync(c2, 0, C2 * sizeof(float), stream);
  for (int k = 0; k < KNN; ++k)
    gemm(stream, feat, CC, 0, 0, nbr + k, KNN,
         PF(I_CPE_CONV_W) + (size_t)k * CC * CC, CC, 0, 0, 0, nullptr, c2, CC,
         0, 0, N_PTS, CC, CC, 1, 1, 1.f, 0, 1);
  k_fuse_bias<<<1, 256, 0, stream>>>(PF(I_CPE_CONV_B), PF(I_CPE_FC1_W),
                                     PF(I_CPE_FC1_B), bias2);
  gemm(stream, c2, CC, 0, 0, nullptr, 0, PF(I_CPE_FC1_W), CC, 0, 0, 1, bias2,
       c3, CC, 0, 0, N_PTS, CC, CC, 1, 1, 1.f, 0, 0);
  k_ln<<<N_PTS, 256, 0, stream>>>(c3, feat, PF(I_CPE_LN_G), PF(I_CPE_LN_B), c4,
                                  1);

  // ======================= Mamba mixer over O orders =======================
  (void)hipMemsetAsync(c5, 0, C2 * sizeof(float), stream);
  for (int i = 0; i < OO; ++i) {
    k_gather<<<N_PTS, 256, 0, stream>>>(c0, c4, inv + (size_t)i * N_PTS);
    for (int b = 0; b < NDEPTH; ++b) {
      const float* src = (b == 0) ? c0 : c2;  // r1 or r2
      k_ln<<<N_PTS, 256, 0, stream>>>(src, nullptr, PF(BLK(b, 0)),
                                      PF(BLK(b, 1)), c1, 0);
      gemm(stream, c1, CC, 0, 0, nullptr, 0, PF(BLK(b, 2)), CC, 0, 0, 1,
           nullptr, big, 2 * DDI, 0, 0, N_PTS, 2 * DDI, CC, 1, 1, 1.f, 0, 0);
      k_conv_silu<<<(N_PTS * DDI) / 256, 256, 0, stream>>>(
          big, PF(BLK(b, 3)), PF(BLK(b, 4)), bid, xcb);
      gemm(stream, xcb, DDI, 0, 0, nullptr, 0, PF(BLK(b, 5)), DDI, 0, 0, 1,
           nullptr, dblb, 48, 0, 0, N_PTS, 48, DDI, 1, 1, 1.f, 0, 0);
      k_dtproj<<<(N_PTS * DDI) / 256, 256, 0, stream>>>(dblb, PF(BLK(b, 6)),
                                                        PF(BLK(b, 7)), dtb);
      k_scan<<<2, 256, 0, stream>>>(dtb, xcb, big, dblb, bid, PF(BLK(b, 8)),
                                    PF(BLK(b, 9)), yb);
      gemm(stream, yb, DDI, 0, 0, nullptr, 0, PF(BLK(b, 10)), DDI, 0, 0, 1,
           nullptr, c3, CC, 0, 0, N_PTS, CC, DDI, 1, 1, 1.f, 0, 0);
      if (b == 0)
        k_add<<<(C2 + 255) / 256, 256, 0, stream>>>(c2, c3, c0, C2);  // r2
      else
        k_add<<<(C2 + 255) / 256, 256, 0, stream>>>(c1, c3, c2, C2);  // h_end
    }
    k_ln<<<N_PTS, 256, 0, stream>>>(c1, nullptr, PF(I_NF_G), PF(I_NF_B), c3, 0);
    k_gather_add<<<N_PTS, 256, 0, stream>>>(c5, c3, order + (size_t)i * N_PTS);
  }
  k_scale<<<(C2 + 255) / 256, 256, 0, stream>>>(c5, 1.f / OO, C2);  // h2

  // ======================= dec2 (cross-only decoder) =======================
  gemm(stream, c5, CC, 0, 0, nullptr, 0, PF(D2 + 0), CC, 0, 0, 1, PF(D2 + 1),
       c6, CC, 0, 0, N_PTS, CC, CC, 1, 1, 1.f, 0, 0);
  gemm(stream, out_query, CC, 0, 0, nullptr, 0, PF(D2 + 2), CC, 0, 0, 1,
       PF(D2 + 3), kq2, CC, 0, 0, BQ, CC, CC, 1, 1, 1.f, 0, 0);
  gemm(stream, out_query, CC, 0, 0, nullptr, 0, PF(D2 + 4), CC, 0, 0, 1,
       PF(D2 + 5), vq2, CC, 0, 0, BQ, CC, CC, 1, 1, 1.f, 0, 0);
  // scores[b,h]: M=G, N=Q, K=DH
  gemm(stream, c6, CC, GC, DH, nullptr, 0, kq2, CC, QC, DH, 1, nullptr, big, QQ,
       (long)HH * GG * QQ, (long)GG * QQ, GG, QQ, DH, BB * HH, HH, rs, 0, 0);
  k_softmax<<<BB * HH * GG, 128, 0, stream>>>(big, QQ);
  gemm(stream, big, QQ, (long)HH * GG * QQ, (long)GG * QQ, nullptr, 0, vq2, CC,
       QC, DH, 0, nullptr, c3, CC, GC, DH, GG, DH, QQ, BB * HH, HH, 1.f, 0, 0);
  gemm(stream, c3, CC, 0, 0, nullptr, 0, PF(D2 + 6), CC, 0, 0, 1, PF(D2 + 7),
       c6, CC, 0, 0, N_PTS, CC, CC, 1, 1, 1.f, 0, 0);
  k_ln<<<N_PTS, 256, 0, stream>>>(c6, c5, PF(D2 + 8), PF(D2 + 9), c4, 2);
  // FFN
  gemm(stream, c4, CC, 0, 0, nullptr, 0, PF(D2 + 12), CC, 0, 0, 1, PF(D2 + 13),
       big, FF, 0, 0, N_PTS, FF, CC, 1, 1, 1.f, 1, 0);
  gemm(stream, big, FF, 0, 0, nullptr, 0, PF(D2 + 14), FF, 0, 0, 1, PF(D2 + 15),
       c6, CC, 0, 0, N_PTS, CC, FF, 1, 1, 1.f, 0, 0);
  k_ln<<<N_PTS, 256, 0, stream>>>(c6, c4, PF(D2 + 10), PF(D2 + 11), out_feat,
                                  2);
}
